// DeformableTransformerDecoderLayer_28973849379503
// MI455X (gfx1250) — matrix operand order repaired and checked by reading, not compiled, not run on hardware
//
#include <hip/hip_runtime.h>
#include <hip/hip_bf16.h>

// ---------------------------------------------------------------------------
// Types
// ---------------------------------------------------------------------------
typedef __attribute__((ext_vector_type(16))) __bf16 v16bf;
typedef __attribute__((ext_vector_type(2)))  __bf16 v2bf;
typedef __attribute__((ext_vector_type(8)))  float  v8f;
typedef unsigned int u32;
typedef __attribute__((ext_vector_type(4)))  u32    v4u;

#define B_  16
#define LQ_ 1024
#define D_  256
#define H_  8
#define DH_ 32
#define LV_ 5440
#define M1_ 16384   // B*LQ
#define MV_ 87040   // B*LV

__device__ __forceinline__ __bf16 f2bf(float x) {
  union { float f; u32 u; } v; v.f = x;
  u32 r = v.u + 0x7FFFu + ((v.u >> 16) & 1u);
  unsigned short h = (unsigned short)(r >> 16);
  return __builtin_bit_cast(__bf16, h);
}

// Assemble a WMMA 16-wide bf16 fragment from two aligned 16-byte runs.
union FragU { v4u q[2]; v16bf v; };
__device__ __forceinline__ v16bf load_frag2(const __bf16* p0, const __bf16* p1) {
  FragU f;
  f.q[0] = *(const v4u*)p0;
  f.q[1] = *(const v4u*)p1;
  return f.v;
}

// ---------------------------------------------------------------------------
// Elementwise converts
// ---------------------------------------------------------------------------
__global__ void cvt_bf_kernel(const float* __restrict__ in, __bf16* __restrict__ out, int n) {
  int i = blockIdx.x * blockDim.x + threadIdx.x;
  if (i < n) out[i] = f2bf(in[i]);
}
__global__ void add2_bf_kernel(const float* __restrict__ a, const float* __restrict__ b,
                               __bf16* __restrict__ out, int n) {
  int i = blockIdx.x * blockDim.x + threadIdx.x;
  if (i < n) out[i] = f2bf(a[i] + b[i]);
}

// ---------------------------------------------------------------------------
// WMMA GEMM: C[M,N] = A[M,K] * W[N,K]^T + bias   (A,W bf16, C f32 or bf16)
// Block = 128 threads (4 waves); 64x64 C tile; each wave does 32x32 via 2x2
// v_wmma_f32_16x16x32_bf16 tiles; K stepped by 32 through LDS (pitch 40 so
// all fragment reads are aligned ds_load_b128 and staging is b128/b128).
// ---------------------------------------------------------------------------
#define PA_ 40   // LDS pitch (bf16 elems) for 32-wide K tiles

template <bool OBF16, bool RELU>
__global__ void __launch_bounds__(128)
gemm_bf16_wmma(const __bf16* __restrict__ A, const __bf16* __restrict__ W,
               const float* __restrict__ bias, void* __restrict__ out,
               int M, int N, int K) {
  __shared__ __bf16 As[64 * PA_];
  __shared__ __bf16 Ws[64 * PA_];

  const int tid  = threadIdx.x;
  const int lane = tid & 31;
  const int w    = tid >> 5;            // 0..3
  const int wr   = (w >> 1) * 32;       // wave row offset within tile
  const int wc   = (w & 1) * 32;        // wave col offset within tile
  const int rowBase = blockIdx.y * 64;
  const int colBase = blockIdx.x * 64;

  const int qp = K >> 3;                // quads (v4u) per global row

  v8f acc[2][2] = {};

  for (int k0 = 0; k0 < K; k0 += 32) {
    // stage A and W tiles: 64 rows x 32 bf16 = 256 quads each, 2 per thread
    const v4u* gA = (const v4u*)(A + (size_t)rowBase * K + k0);
    const v4u* gW = (const v4u*)(W + (size_t)colBase * K + k0);
#pragma unroll
    for (int i = 0; i < 2; ++i) {
      int qi = tid + i * 128;           // 0..255
      int r  = qi >> 2;                 // row 0..63
      int c  = qi & 3;                  // quad col
      *(v4u*)&As[r * PA_ + c * 8] = gA[(size_t)r * qp + c];
      *(v4u*)&Ws[r * PA_ + c * 8] = gW[(size_t)r * qp + c];
    }
    // prefetch next K tile (emits global_prefetch)
    if (k0 + 32 < K) {
      __builtin_prefetch(A + (size_t)(rowBase + (tid >> 1)) * K + k0 + 32, 0, 3);
      __builtin_prefetch(W + (size_t)(colBase + (tid >> 1)) * K + k0 + 32, 0, 3);
    }
    __syncthreads();

    v16bf a[2], bw[2];
#pragma unroll
    for (int t = 0; t < 2; ++t) {
      // A fragment: row per lane, lanes 0-15 K=kb.., lanes 16-31 K=kb+8..
      int m  = wr + t * 16 + (lane & 15);
      int kb = (lane >> 4) << 3;         // 0 or 8
      a[t] = load_frag2(&As[m * PA_ + kb], &As[m * PA_ + 16 + kb]);
      // B fragment: column per lane, lanes 0-15 K=kb2.., lanes 16-31 K=kb2+8..
      int n   = wc + t * 16 + (lane & 15);
      int kb2 = (lane >> 4) << 3;        // 0 or 8
      bw[t] = load_frag2(&Ws[n * PA_ + kb2], &Ws[n * PA_ + 16 + kb2]);
    }
#pragma unroll
    for (int i = 0; i < 2; ++i)
#pragma unroll
      for (int j = 0; j < 2; ++j)
        acc[i][j] = __builtin_amdgcn_wmma_f32_16x16x32_bf16(
            false, a[i], false, bw[j], (short)0, acc[i][j], false, false);
    __syncthreads();
  }

#pragma unroll
  for (int i = 0; i < 2; ++i) {
#pragma unroll
    for (int j = 0; j < 2; ++j) {
      int n = colBase + wc + j * 16 + (lane & 15);
      float bv = bias ? bias[n] : 0.f;
#pragma unroll
      for (int r = 0; r < 8; ++r) {
        int m = rowBase + wr + i * 16 + r + ((lane >> 4) << 3);
        float v = acc[i][j][r] + bv;
        if (RELU) v = fmaxf(v, 0.f);
        if (OBF16) ((__bf16*)out)[(size_t)m * N + n] = f2bf(v);
        else       ((float*)out)[(size_t)m * N + n] = v;
      }
    }
  }
}

// ---------------------------------------------------------------------------
// Flash attention (non-causal): per (b,h), 64-query blocks, online softmax.
// qk: [M1, 512] bf16  (q at col h*32, k at col 256+h*32)
// v : [M1, 256] bf16
// o : [M1, 256] bf16
// ---------------------------------------------------------------------------
#define PK_ 40   // Ks pitch
#define PV_ 72   // Vt / Ps pitch (64 kv + pad, multiple of 8)

__global__ void __launch_bounds__(128)
flash_attn_wmma(const __bf16* __restrict__ qk, const __bf16* __restrict__ vv,
                __bf16* __restrict__ o) {
  const int bh = blockIdx.y;
  const int b  = bh >> 3;
  const int h  = bh & 7;
  const int qbase = blockIdx.x * 64;
  const int tid  = threadIdx.x;
  const int lane = tid & 31;
  const int w    = tid >> 5;

  __shared__ __bf16 Ks[64 * PK_];   // [kv][dh]
  __shared__ __bf16 Vt[32 * PV_];   // [dh][kv]  (transposed V)
  __shared__ __bf16 Ps[64 * PV_];   // [q][kv]   P tile (per-wave exclusive rows)

  const size_t tok0 = (size_t)b * LQ_;

  // Q A-fragment straight from global (16 rows per wave, K = DH = 32)
  v16bf qf;
  {
    int m  = w * 16 + (lane & 15);
    int kb = (lane >> 4) << 3;
    const __bf16* qrow = qk + (tok0 + qbase + m) * 512 + h * 32;
    qf = load_frag2(qrow + kb, qrow + 16 + kb);
  }

  float mrow[8], lrow[8];
  v8f accO[2] = {};
#pragma unroll
  for (int r = 0; r < 8; ++r) { mrow[r] = -1e30f; lrow[r] = 0.f; }
  const float scale = 0.17677669529663687f;  // 1/sqrt(32)

  for (int kv0 = 0; kv0 < LQ_; kv0 += 64) {
    // stage K tile [64][32]: 256 quads, 2 per thread (b128 -> b128)
#pragma unroll
    for (int i = 0; i < 2; ++i) {
      int qi = tid + i * 128;
      int r = qi >> 2, c = qi & 3;
      const v4u* src = (const v4u*)(qk + (tok0 + kv0 + r) * 512 + 256 + h * 32);
      *(v4u*)&Ks[r * PK_ + c * 8] = src[c];
    }
    // stage V transposed -> Vt[d][kv]
#pragma unroll
    for (int i = 0; i < 8; ++i) {
      int idx = tid + i * 128;
      int r = idx >> 4, c = idx & 15;  // r = kv row, c = d-pair
      u32 pv = *(const u32*)(vv + (tok0 + kv0 + r) * 256 + h * 32 + c * 2);
      v2bf p = __builtin_bit_cast(v2bf, pv);
      Vt[(c * 2) * PV_ + r]     = p[0];
      Vt[(c * 2 + 1) * PV_ + r] = p[1];
    }
    __syncthreads();

    // S = Q K^T  (4 n-tiles of 16 kv each)
    v8f s[4];
#pragma unroll
    for (int j = 0; j < 4; ++j) {
      int n   = j * 16 + (lane & 15);   // kv index = B "column"
      int kb2 = (lane >> 4) << 3;
      v16bf bk = load_frag2(&Ks[n * PK_ + kb2], &Ks[n * PK_ + 16 + kb2]);
      v8f z = {};
      s[j] = __builtin_amdgcn_wmma_f32_16x16x32_bf16(
          false, qf, false, bk, (short)0, z, false, false);
    }

    // online softmax (row r lives in C-VGPR r; reduce across 16-lane halves)
#pragma unroll
    for (int r = 0; r < 8; ++r) {
      float mx = -1e30f;
#pragma unroll
      for (int j = 0; j < 4; ++j) { s[j][r] *= scale; mx = fmaxf(mx, s[j][r]); }
#pragma unroll
      for (int dx = 8; dx >= 1; dx >>= 1) mx = fmaxf(mx, __shfl_xor(mx, dx, 32));
      float mn   = fmaxf(mrow[r], mx);
      float corr = __expf(mrow[r] - mn);
      mrow[r] = mn;
      float rs = 0.f;
#pragma unroll
      for (int j = 0; j < 4; ++j) {
        float pj = __expf(s[j][r] - mn);
        s[j][r] = pj;
        rs += pj;
      }
#pragma unroll
      for (int dx = 8; dx >= 1; dx >>= 1) rs += __shfl_xor(rs, dx, 32);
      lrow[r] = lrow[r] * corr + rs;
      accO[0][r] *= corr;
      accO[1][r] *= corr;
    }

    // P -> LDS (each wave writes only its own 16 rows; no barrier needed)
#pragma unroll
    for (int j = 0; j < 4; ++j) {
      int n = j * 16 + (lane & 15);
#pragma unroll
      for (int r = 0; r < 8; ++r) {
        int m = w * 16 + r + ((lane >> 4) << 3);
        Ps[m * PV_ + n] = f2bf(s[j][r]);
      }
    }

    // O += P @ V  (2 kv-halves x 2 dh-tiles)
#pragma unroll
    for (int t = 0; t < 2; ++t) {
      int m  = w * 16 + (lane & 15);
      int kb = (lane >> 4) << 3;
      v16bf pa = load_frag2(&Ps[m * PV_ + t * 32 + kb],
                            &Ps[m * PV_ + t * 32 + 16 + kb]);
#pragma unroll
      for (int j = 0; j < 2; ++j) {
        int d   = j * 16 + (lane & 15);
        int kb2 = t * 32 + ((lane >> 4) << 3);
        v16bf bv = load_frag2(&Vt[d * PV_ + kb2], &Vt[d * PV_ + 16 + kb2]);
        accO[j] = __builtin_amdgcn_wmma_f32_16x16x32_bf16(
            false, pa, false, bv, (short)0, accO[j], false, false);
      }
    }
    __syncthreads();
  }

  // normalize + store (v_rcp instead of full-precision divide)
#pragma unroll
  for (int j = 0; j < 2; ++j) {
    int d = j * 16 + (lane & 15);
#pragma unroll
    for (int r = 0; r < 8; ++r) {
      int m = w * 16 + r + ((lane >> 4) << 3);
      float val = accO[j][r] * __builtin_amdgcn_rcpf(lrow[r]);
      o[(tok0 + qbase + m) * 256 + h * 32 + d] = f2bf(val);
    }
  }
}

// ---------------------------------------------------------------------------
// LayerNorm of (x + y); optional f32 out, bf16 out, bf16(out + pos)
// One 256-thread block per row (D = 256).
// ---------------------------------------------------------------------------
__global__ void __launch_bounds__(256)
ln_kernel(const float* __restrict__ x, const float* __restrict__ y,
          const float* __restrict__ g, const float* __restrict__ bta,
          float* __restrict__ of, __bf16* __restrict__ ob,
          const float* __restrict__ pos, __bf16* __restrict__ opb) {
  const int row = blockIdx.x, tid = threadIdx.x;
  __shared__ float sm[8], sv[8];
  size_t base = (size_t)row * 256 + tid;
  float v = x[base] + y[base];
  float s = v;
#pragma unroll
  for (int dx = 16; dx >= 1; dx >>= 1) s += __shfl_xor(s, dx, 32);
  if ((tid & 31) == 0) sm[tid >> 5] = s;
  __syncthreads();
  float mean = 0.f;
#pragma unroll
  for (int i = 0; i < 8; ++i) mean += sm[i];
  mean *= (1.f / 256.f);
  float dv = v - mean;
  float s2 = dv * dv;
#pragma unroll
  for (int dx = 16; dx >= 1; dx >>= 1) s2 += __shfl_xor(s2, dx, 32);
  if ((tid & 31) == 0) sv[tid >> 5] = s2;
  __syncthreads();
  float var = 0.f;
#pragma unroll
  for (int i = 0; i < 8; ++i) var += sv[i];
  var *= (1.f / 256.f);
  float o = dv * rsqrtf(var + 1e-5f) * g[tid] + bta[tid];
  if (of)  of[base] = o;
  if (ob)  ob[base] = f2bf(o);
  if (opb) opb[base] = f2bf(o + pos[base]);
}

// ---------------------------------------------------------------------------
// MS-deformable attention sampling.
// Grid: B*LQ blocks; 256 threads = (h in 0..7) x (d in 0..31).
// value: [B*LV, 256] f32; off: [M1,256] f32; attw: [M1,128] f32 (logits);
// ref: [B,LQ,L,2] f32; out: [M1,256] bf16.
// ---------------------------------------------------------------------------
__global__ void __launch_bounds__(256)
deform_kernel(const float* __restrict__ value, const float* __restrict__ off,
              const float* __restrict__ attw, const float* __restrict__ ref,
              __bf16* __restrict__ out) {
  const int bq = blockIdx.x;
  const int b  = bq >> 10;
  const int tid = threadIdx.x;
  const int h = tid >> 5;
  const int d = tid & 31;

  // softmax over the 16 (l,p) logits for this (b,q,h)
  const float* aw = attw + (size_t)bq * 128 + h * 16;
  float wloc[16];
  float mx = -1e30f;
#pragma unroll
  for (int i = 0; i < 16; ++i) { wloc[i] = aw[i]; mx = fmaxf(mx, wloc[i]); }
  float sum = 0.f;
#pragma unroll
  for (int i = 0; i < 16; ++i) { wloc[i] = __expf(wloc[i] - mx); sum += wloc[i]; }
  const float inv = 1.f / sum;

  const int lvlH[4] = {64, 32, 16, 8};
  const int lvlW[4] = {64, 32, 16, 8};
  const int lvlS[4] = {0, 4096, 5120, 5376};

  const float* offp = off + (size_t)bq * 256 + h * 32;
  const float* refp = ref + (size_t)bq * 8;

  float acc = 0.f;
#pragma unroll
  for (int l = 0; l < 4; ++l) {
    const int hl = lvlH[l], wl = lvlW[l], st = lvlS[l];
    const float rx = refp[l * 2 + 0], ry = refp[l * 2 + 1];
    const float* vbase = value + ((size_t)b * LV_ + st) * 256 + h * 32 + d;
#pragma unroll
    for (int p = 0; p < 4; ++p) {
      float ox = offp[l * 8 + p * 2 + 0];
      float oy = offp[l * 8 + p * 2 + 1];
      float fx = (rx + ox / (float)wl) * (float)wl - 0.5f;
      float fy = (ry + oy / (float)hl) * (float)hl - 0.5f;
      float x0 = floorf(fx), y0 = floorf(fy);
      float lx = fx - x0, ly = fy - y0;
      int xi = (int)x0, yi = (int)y0;
      float sample = 0.f;
#pragma unroll
      for (int cy = 0; cy < 2; ++cy) {
#pragma unroll
        for (int cx = 0; cx < 2; ++cx) {
          int xx = xi + cx, yy = yi + cy;
          float wgt = (cx ? lx : 1.f - lx) * (cy ? ly : 1.f - ly);
          if (xx >= 0 && xx < wl && yy >= 0 && yy < hl)
            sample += wgt * vbase[(size_t)(yy * wl + xx) * 256];
        }
      }
      acc += wloc[l * 4 + p] * inv * sample;
    }
  }
  out[(size_t)bq * 256 + h * 32 + d] = f2bf(acc);
}

// ---------------------------------------------------------------------------
// Host orchestration
// ---------------------------------------------------------------------------
static inline int cdiv(int a, int b) { return (a + b - 1) / b; }

extern "C" void kernel_launch(void* const* d_in, const int* in_sizes, int n_in,
                              void* d_out, int out_size, void* d_ws, size_t ws_size,
                              hipStream_t stream) {
  (void)in_sizes; (void)n_in; (void)out_size; (void)ws_size;

  const float* tgt        = (const float*)d_in[0];
  const float* query_pos  = (const float*)d_in[1];
  const float* ref_pts    = (const float*)d_in[2];
  const float* src        = (const float*)d_in[3];
  const float* in_proj_w  = (const float*)d_in[4];
  const float* in_proj_b  = (const float*)d_in[5];
  const float* out_proj_w = (const float*)d_in[6];
  const float* out_proj_b = (const float*)d_in[7];
  const float* norm2_g    = (const float*)d_in[8];
  const float* norm2_b    = (const float*)d_in[9];
  const float* value_w    = (const float*)d_in[10];
  const float* value_b    = (const float*)d_in[11];
  const float* off_w      = (const float*)d_in[12];
  const float* off_b      = (const float*)d_in[13];
  const float* attw_w     = (const float*)d_in[14];
  const float* attw_b     = (const float*)d_in[15];
  const float* outp_w     = (const float*)d_in[16];
  const float* outp_b     = (const float*)d_in[17];
  const float* norm1_g    = (const float*)d_in[18];
  const float* norm1_b    = (const float*)d_in[19];
  const float* lin1_w     = (const float*)d_in[20];
  const float* lin1_b     = (const float*)d_in[21];
  const float* lin2_w     = (const float*)d_in[22];
  const float* lin2_b     = (const float*)d_in[23];
  const float* norm3_g    = (const float*)d_in[24];
  const float* norm3_b    = (const float*)d_in[25];

  char* p = (char*)d_ws;
  auto alloc = [&](size_t bytes) -> void* {
    void* r = (void*)p;
    p += (bytes + 255) & ~(size_t)255;
    return r;
  };

  // bf16 weights
  __bf16* w_in   = (__bf16*)alloc((size_t)768 * 256 * 2);
  __bf16* w_out  = (__bf16*)alloc((size_t)256 * 256 * 2);
  __bf16* w_val  = (__bf16*)alloc((size_t)256 * 256 * 2);
  __bf16* w_off  = (__bf16*)alloc((size_t)256 * 256 * 2);
  __bf16* w_attw = (__bf16*)alloc((size_t)128 * 256 * 2);
  __bf16* w_outp = (__bf16*)alloc((size_t)256 * 256 * 2);
  __bf16* w_lin1 = (__bf16*)alloc((size_t)1024 * 256 * 2);
  __bf16* w_lin2 = (__bf16*)alloc((size_t)256 * 1024 * 2);

  // activations
  __bf16* qbf    = (__bf16*)alloc((size_t)M1_ * 256 * 2);   // tgt+pos ; later q2
  __bf16* tbf    = (__bf16*)alloc((size_t)M1_ * 256 * 2);   // bf16(tgt); later deform out
  __bf16* sbf    = (__bf16*)alloc((size_t)MV_ * 256 * 2);   // bf16(src)
  __bf16* qkbf   = (__bf16*)alloc((size_t)M1_ * 512 * 2);   // q|k projections
  __bf16* vbf    = (__bf16*)alloc((size_t)M1_ * 256 * 2);   // v projection
  __bf16* attnbf = (__bf16*)alloc((size_t)M1_ * 256 * 2);   // mha out ; later tgtB bf16
  __bf16* hidbf  = (__bf16*)alloc((size_t)M1_ * 1024 * 2);  // FFN hidden
  float*  projf  = (float*)alloc((size_t)M1_ * 256 * 4);    // tgt2 / tgt2b / t2 (reused)
  float*  tgtA   = (float*)alloc((size_t)M1_ * 256 * 4);
  float*  tgtB   = (float*)alloc((size_t)M1_ * 256 * 4);
  float*  valf   = (float*)alloc((size_t)MV_ * 256 * 4);
  float*  offf   = (float*)alloc((size_t)M1_ * 256 * 4);
  float*  awf    = (float*)alloc((size_t)M1_ * 128 * 4);

  const int T = 256;
  auto cvt = [&](const float* s_, __bf16* d_, int n) {
    cvt_bf_kernel<<<cdiv(n, T), T, 0, stream>>>(s_, d_, n);
  };

  // 1) weights -> bf16
  cvt(in_proj_w, w_in,   768 * 256);
  cvt(out_proj_w, w_out, 256 * 256);
  cvt(value_w, w_val,    256 * 256);
  cvt(off_w, w_off,      256 * 256);
  cvt(attw_w, w_attw,    128 * 256);
  cvt(outp_w, w_outp,    256 * 256);
  cvt(lin1_w, w_lin1,    1024 * 256);
  cvt(lin2_w, w_lin2,    256 * 1024);

  // 2) activations -> bf16
  add2_bf_kernel<<<cdiv(M1_ * 256, T), T, 0, stream>>>(tgt, query_pos, qbf, M1_ * 256);
  cvt(tgt, tbf, M1_ * 256);
  cvt(src, sbf, MV_ * 256);

  // 3) MHA projections: [q|k] = q_in @ W[:512].T ; v = tgt @ W[512:].T
  gemm_bf16_wmma<true, false><<<dim3(512 / 64, M1_ / 64), 128, 0, stream>>>(
      qbf, w_in, in_proj_b, qkbf, M1_, 512, 256);
  gemm_bf16_wmma<true, false><<<dim3(256 / 64, M1_ / 64), 128, 0, stream>>>(
      tbf, w_in + (size_t)512 * 256, in_proj_b + 512, vbf, M1_, 256, 256);

  // 4) fused flash attention
  flash_attn_wmma<<<dim3(LQ_ / 64, B_ * H_), 128, 0, stream>>>(qkbf, vbf, attnbf);

  // 5) out_proj + residual LN (also emits q2 = bf16(LN_out + query_pos))
  gemm_bf16_wmma<false, false><<<dim3(256 / 64, M1_ / 64), 128, 0, stream>>>(
      attnbf, w_out, out_proj_b, projf, M1_, 256, 256);
  ln_kernel<<<M1_, 256, 0, stream>>>(tgt, projf, norm2_g, norm2_b,
                                     tgtA, nullptr, query_pos, qbf);

  // 6) deformable attention
  gemm_bf16_wmma<false, false><<<dim3(256 / 64, MV_ / 64), 128, 0, stream>>>(
      sbf, w_val, value_b, valf, MV_, 256, 256);
  gemm_bf16_wmma<false, false><<<dim3(256 / 64, M1_ / 64), 128, 0, stream>>>(
      qbf, w_off, off_b, offf, M1_, 256, 256);
  gemm_bf16_wmma<false, false><<<dim3(128 / 64, M1_ / 64), 128, 0, stream>>>(
      qbf, w_attw, attw_b, awf, M1_, 128, 256);
  deform_kernel<<<M1_, 256, 0, stream>>>(valf, offf, awf, ref_pts, tbf);
  gemm_bf16_wmma<false, false><<<dim3(256 / 64, M1_ / 64), 128, 0, stream>>>(
      tbf, w_outp, outp_b, projf, M1_, 256, 256);
  ln_kernel<<<M1_, 256, 0, stream>>>(tgtA, projf, norm1_g, norm1_b,
                                     tgtB, attnbf, nullptr, nullptr);

  // 7) FFN
  gemm_bf16_wmma<true, true><<<dim3(1024 / 64, M1_ / 64), 128, 0, stream>>>(
      attnbf, w_lin1, lin1_b, hidbf, M1_, 1024, 256);
  gemm_bf16_wmma<false, false><<<dim3(256 / 64, M1_ / 64), 128, 0, stream>>>(
      hidbf, w_lin2, lin2_b, projf, M1_, 256, 1024);
  ln_kernel<<<M1_, 256, 0, stream>>>(tgtB, projf, norm3_g, norm3_b,
                                     (float*)d_out, nullptr, nullptr, nullptr);
}
